// GNNLayerAttention_36850819400035
// MI455X (gfx1250) — matrix operand order repaired and checked by reading, not compiled, not run on hardware
//
#include <hip/hip_runtime.h>
#include <hip/hip_bf16.h>

typedef __attribute__((ext_vector_type(16))) _Float16 v16h;
typedef __attribute__((ext_vector_type(8)))  _Float16 v8h;
typedef __attribute__((ext_vector_type(8)))  float    v8f;
typedef __attribute__((ext_vector_type(4)))  float    v4f;

#define NEG_SLOPE 0.01f

__device__ __forceinline__ float leakyf(float x) { return x >= 0.f ? x : NEG_SLOPE * x; }

__device__ __forceinline__ void atomicAddF(float* p, float v) {
    (void)__hip_atomic_fetch_add(p, v, __ATOMIC_RELAXED, __HIP_MEMORY_SCOPE_AGENT);
}

// Native CDNA5 fp32 global atomic add, device scope, no return (STOREcnt path).
__device__ __forceinline__ void atomicAddAsm(float* p, float v) {
    asm volatile("global_atomic_add_f32 %0, %1, off scope:SCOPE_DEV"
                 :: "v"(p), "v"(v)
                 : "memory");
}

// ---------------------------------------------------------------------------
// Kernel 0a: fold attention:  w_s = W_att.T @ a[:128], w_d = W_att.T @ a[128:],
//            c = W_att_b · (a[:128] + a[128:])         (tiny: 128x128)
// wsv layout: [0..127]=w_s  [128..255]=w_d  [256]=c
// ---------------------------------------------------------------------------
__global__ __launch_bounds__(128)
void k_att_fold(const float* __restrict__ Waw, const float* __restrict__ Wab,
                const float* __restrict__ a, float* __restrict__ wsv)
{
    int k = threadIdx.x;  // 0..127
    float s1 = 0.f, s2 = 0.f;
    for (int n = 0; n < 128; ++n) {
        float w = Waw[n * 128 + k];
        s1 += w * a[n];
        s2 += w * a[128 + n];
    }
    wsv[k] = s1;
    wsv[128 + k] = s2;
    if (k == 0) {
        float c = 0.f;
        for (int n = 0; n < 128; ++n) c += Wab[n] * (a[n] + a[128 + n]);
        wsv[256] = c;
    }
}

// ---------------------------------------------------------------------------
// Kernel 0b: one-shot f32 -> f16 conversion of W1 and W2 (row-major kept)
// ---------------------------------------------------------------------------
__global__ __launch_bounds__(256)
void k_wconv(const float* __restrict__ W1w, const float* __restrict__ W2w,
             _Float16* __restrict__ w1h, _Float16* __restrict__ w2h)
{
    int i = blockIdx.x * blockDim.x + threadIdx.x;
    if (i < 128 * 128) {
        w1h[i] = (_Float16)W1w[i];
        w2h[i] = (_Float16)W2w[i];
    }
}

// Build a WMMA B-fragment for chunk c from pre-converted f16 weights.
// Lane (n = L%16, h = L/16) holds W[n][c*32 + h*8 + 0..7] and W[n][c*32 + 16 + h*8 + 0..7]  (=> B[k][n] = W[n][k]).
__device__ __forceinline__ v16h load_B(const _Float16* __restrict__ wrow, int c, int h)
{
    int kb = c * 32 + h * 8;
    v8h lo = *(const v8h*)(wrow + kb);
    v8h hi = *(const v8h*)(wrow + kb + 16);
    v16h vb;
#pragma unroll
    for (int i = 0; i < 8; ++i) {
        vb[i]     = lo[i];
        vb[8 + i] = hi[i];
    }
    return vb;
}

// ---------------------------------------------------------------------------
// Kernel 1: per 16-row node tile (one wave each):
//   h_msg = feat @ W1.T + W1_b      (WMMA f16, f32 accum; stored f16)
//   e_src = feat · w_s ; e_dst = feat · w_d + c   (rides on the f32 A loads)
// Assumes N % 16 == 0 (N = 100000 = 6250*16).
// ---------------------------------------------------------------------------
__global__ __launch_bounds__(128)
void k_node_gemm(const float* __restrict__ feat,
                 const _Float16* __restrict__ w1h, const float* __restrict__ W1b,
                 const float* __restrict__ wsv,
                 _Float16* __restrict__ hmsg, float* __restrict__ esrc,
                 float* __restrict__ edst, int ntiles)
{
    int wave = (int)((blockIdx.x * blockDim.x + threadIdx.x) >> 5);
    if (wave >= ntiles) return;
    int lane = threadIdx.x & 31;
    int m = lane & 15;       // row within tile (A layout), also col within n-tile (B/D)
    int h = lane >> 4;       // lane half
    long nb = (long)wave * 16;
    const float* rowp = feat + (nb + m) * 128;

    v16h A[4];
    float ds = 0.f, dd = 0.f;
    const float cconst = wsv[256];

#pragma unroll
    for (int c = 0; c < 4; ++c) {
        int k1 = c * 32 + h * 8;          // elements [0..7]  -> k1..k1+7
        // elements [8..15] -> k1+16..k1+23
        v4f f0 = *(const v4f*)(rowp + k1);
        v4f f1 = *(const v4f*)(rowp + k1 + 4);
        v4f f2 = *(const v4f*)(rowp + k1 + 16);
        v4f f3 = *(const v4f*)(rowp + k1 + 20);
        v4f s0 = *(const v4f*)(wsv + k1);
        v4f s1 = *(const v4f*)(wsv + k1 + 4);
        v4f s2 = *(const v4f*)(wsv + k1 + 16);
        v4f s3 = *(const v4f*)(wsv + k1 + 20);
        v4f d0 = *(const v4f*)(wsv + 128 + k1);
        v4f d1 = *(const v4f*)(wsv + 128 + k1 + 4);
        v4f d2 = *(const v4f*)(wsv + 128 + k1 + 16);
        v4f d3 = *(const v4f*)(wsv + 128 + k1 + 20);
        v16h va;
#pragma unroll
        for (int i = 0; i < 4; ++i) {
            ds += f0[i] * s0[i] + f1[i] * s1[i] + f2[i] * s2[i] + f3[i] * s3[i];
            dd += f0[i] * d0[i] + f1[i] * d1[i] + f2[i] * d2[i] + f3[i] * d3[i];
            va[i]      = (_Float16)f0[i];
            va[4 + i]  = (_Float16)f1[i];
            va[8 + i]  = (_Float16)f2[i];
            va[12 + i] = (_Float16)f3[i];
        }
        A[c] = va;
    }

    // GEMM: 8 n-tiles of 16 cols, K=128 in 4 chunks
#pragma unroll
    for (int nt = 0; nt < 8; ++nt) {
        int n = nt * 16 + m;             // global output column
        float bn = W1b[n];
        v8f acc;
#pragma unroll
        for (int r = 0; r < 8; ++r) acc[r] = bn;
        const _Float16* wr = w1h + n * 128;
#pragma unroll
        for (int c = 0; c < 4; ++c) {
            acc = __builtin_amdgcn_wmma_f32_16x16x32_f16(
                false, A[c], false, load_B(wr, c, h), (short)0, acc, false, false);
        }
#pragma unroll
        for (int r = 0; r < 8; ++r) {
            hmsg[(nb + h * 8 + r) * 128 + n] = (_Float16)acc[r];
        }
    }

    // reduce attention dots across the two lane-halves of each row
    ds += __shfl_xor(ds, 16, 32);
    dd += __shfl_xor(dd, 16, 32);
    if (h == 0) {
        esrc[nb + m] = ds;
        edst[nb + m] = dd + cconst;
    }
}

// ---------------------------------------------------------------------------
// Kernel 2: per edge: exp(leaky(e_src[src]+e_dst[dst])), segment-sum denom
// ---------------------------------------------------------------------------
__global__ __launch_bounds__(256)
void k_edge_logits(const int* __restrict__ src, const int* __restrict__ dst,
                   const float* __restrict__ esrc, const float* __restrict__ edst,
                   float* __restrict__ expl, float* __restrict__ denom, int E)
{
    int e = blockIdx.x * blockDim.x + threadIdx.x;
    if (e >= E) return;
    int s = src[e], d = dst[e];
    float ex = expf(leakyf(esrc[s] + edst[d]));
    expl[e] = ex;
    atomicAddF(denom + d, ex);
}

// ---------------------------------------------------------------------------
// Kernel 3: one wave per edge: h_neigh[dst] += alpha * h_msg[src]  (128 cols)
// Native global_atomic_add_f32 (device scope, no return) via inline asm.
// ---------------------------------------------------------------------------
__global__ __launch_bounds__(256)
void k_scatter(const int* __restrict__ src, const int* __restrict__ dst,
               const float* __restrict__ expl, const float* __restrict__ denom,
               const _Float16* __restrict__ hmsg, float* __restrict__ hneigh, int E)
{
    int wave = (int)((blockIdx.x * blockDim.x + threadIdx.x) >> 5);
    if (wave >= E) return;
    int lane = threadIdx.x & 31;
    int s = src[wave], d = dst[wave];
    float alpha = expl[wave] / (denom[d] + 1e-9f);
    const _Float16* sp = hmsg + (long)s * 128;
    float* dp = hneigh + (long)d * 128;
#pragma unroll
    for (int j = 0; j < 4; ++j) {
        float v = alpha * (float)sp[j * 32 + lane];
        atomicAddAsm(dp + j * 32 + lane, v);
    }
}

// ---------------------------------------------------------------------------
// Kernel 4: out = leaky( feat + h_neigh + (feat*h_neigh) @ W2.T + W2_b )
// h_neigh lives in d_out; updated in-place (each wave owns its 16 rows).
// ---------------------------------------------------------------------------
__global__ __launch_bounds__(128)
void k_final(const float* __restrict__ feat,
             const _Float16* __restrict__ w2h, const float* __restrict__ W2b,
             float* __restrict__ out, int ntiles)
{
    int wave = (int)((blockIdx.x * blockDim.x + threadIdx.x) >> 5);
    if (wave >= ntiles) return;
    int lane = threadIdx.x & 31;
    int m = lane & 15;
    int h = lane >> 4;
    long nb = (long)wave * 16;
    const float* fp = feat + (nb + m) * 128;
    const float* hp = out + (nb + m) * 128;

    v16h A[4];
#pragma unroll
    for (int c = 0; c < 4; ++c) {
        int k1 = c * 32 + h * 8;
        v4f f0 = *(const v4f*)(fp + k1);
        v4f f1 = *(const v4f*)(fp + k1 + 4);
        v4f f2 = *(const v4f*)(fp + k1 + 16);
        v4f f3 = *(const v4f*)(fp + k1 + 20);
        v4f g0 = *(const v4f*)(hp + k1);
        v4f g1 = *(const v4f*)(hp + k1 + 4);
        v4f g2 = *(const v4f*)(hp + k1 + 16);
        v4f g3 = *(const v4f*)(hp + k1 + 20);
        v16h va;
#pragma unroll
        for (int i = 0; i < 4; ++i) {
            va[i]      = (_Float16)(f0[i] * g0[i]);
            va[4 + i]  = (_Float16)(f1[i] * g1[i]);
            va[8 + i]  = (_Float16)(f2[i] * g2[i]);
            va[12 + i] = (_Float16)(f3[i] * g3[i]);
        }
        A[c] = va;
    }

#pragma unroll
    for (int nt = 0; nt < 8; ++nt) {
        int n = nt * 16 + m;
        float bn = W2b[n];
        v8f acc;
#pragma unroll
        for (int r = 0; r < 8; ++r) acc[r] = bn;
        const _Float16* wr = w2h + n * 128;
#pragma unroll
        for (int c = 0; c < 4; ++c) {
            acc = __builtin_amdgcn_wmma_f32_16x16x32_f16(
                false, A[c], false, load_B(wr, c, h), (short)0, acc, false, false);
        }
#pragma unroll
        for (int r = 0; r < 8; ++r) {
            long row = nb + h * 8 + r;
            float fv = feat[row * 128 + n];
            float hv = out[row * 128 + n];
            out[row * 128 + n] = leakyf(fv + hv + acc[r]);
        }
    }
}

// ---------------------------------------------------------------------------
extern "C" void kernel_launch(void* const* d_in, const int* in_sizes, int n_in,
                              void* d_out, int out_size, void* d_ws, size_t ws_size,
                              hipStream_t stream)
{
    const int*   idx  = (const int*)d_in[0];
    const float* feat = (const float*)d_in[1];
    // d_in[2] = num_nodes (device scalar, unused; N derived from in_sizes)
    const float* Waw  = (const float*)d_in[3];
    const float* Wab  = (const float*)d_in[4];
    const float* W1w  = (const float*)d_in[5];
    const float* W1b  = (const float*)d_in[6];
    const float* W2w  = (const float*)d_in[7];
    const float* W2b  = (const float*)d_in[8];
    const float* a    = (const float*)d_in[9];

    int E = in_sizes[0] / 2;
    int N = in_sizes[1] / 128;
    const int* src = idx;
    const int* dst = idx + E;

    // workspace layout
    float* ws       = (float*)d_ws;
    _Float16* hmsg  = (_Float16*)ws;          // N*128 halves  (== N*64 floats)
    float* esrc     = ws + (size_t)N * 64;    // N
    float* edst     = esrc + N;               // N
    float* denom    = edst + N;               // N
    float* wsv      = denom + N;              // 260 (w_s, w_d, c)
    float* expl     = wsv + 260;              // E
    _Float16* w1h   = (_Float16*)(expl + E);  // 16384 halves
    _Float16* w2h   = w1h + 128 * 128;        // 16384 halves
    float* out      = (float*)d_out;

    hipMemsetAsync(denom, 0, (size_t)N * sizeof(float), stream);
    hipMemsetAsync(out, 0, (size_t)N * 128 * sizeof(float), stream);

    k_att_fold<<<1, 128, 0, stream>>>(Waw, Wab, a, wsv);
    k_wconv<<<(128 * 128 + 255) / 256, 256, 0, stream>>>(W1w, W2w, w1h, w2h);

    int ntiles = (N + 15) / 16;               // N%16==0 for this problem
    dim3 gblk((ntiles + 3) / 4);              // 4 waves (tiles) per 128-thread block
    k_node_gemm<<<gblk, 128, 0, stream>>>(feat, w1h, W1b, wsv, hmsg, esrc, edst, ntiles);

    k_edge_logits<<<(E + 255) / 256, 256, 0, stream>>>(src, dst, esrc, edst, expl, denom, E);

    k_scatter<<<(E + 7) / 8, 256, 0, stream>>>(src, dst, expl, denom, hmsg, out, E);

    k_final<<<gblk, 128, 0, stream>>>(feat, W2b ? w2h : w2h, W2b, out, ntiles);
}
